// FHETinyGPT_7679401526021
// MI455X (gfx1250) — matrix operand order repaired and checked by reading, not compiled, not run on hardware
//
#include <hip/hip_runtime.h>

// ---------------------------------------------------------------------------
// FHE TinyGPT forward on gfx1250 (CDNA5): all GEMMs via v_wmma_f32_16x16x32_f16
// Double-buffered LDS pipeline; async global->LDS staging (ASYNCcnt path).
// ---------------------------------------------------------------------------

typedef __attribute__((ext_vector_type(16))) _Float16 v16h;
typedef __attribute__((ext_vector_type(8)))  _Float16 v8h;
typedef __attribute__((ext_vector_type(8)))  float    v8f;
typedef __attribute__((ext_vector_type(4)))  float    v4f;

// exact pointee type expected by __builtin_amdgcn_global_load_async_to_lds_b128
typedef int v4i_vs __attribute__((vector_size(4 * sizeof(int))));
typedef __attribute__((address_space(1))) v4i_vs glb_b128;
typedef __attribute__((address_space(3))) v4i_vs lds_b128;

#define BM   128
#define BN   128
#define BKT  64     // K tile (two wmma k-steps of 32)
#define LDT  72     // padded LDS row stride in halfs (144B, 16B aligned)

#if __has_builtin(__builtin_amdgcn_global_load_async_to_lds_b128)
#define FHE_ASYNC_LDS 1
#define ASYNC_B128(gp, lp) \
    __builtin_amdgcn_global_load_async_to_lds_b128((glb_b128*)(gp), (lds_b128*)(lp), 0, 0)
#else
#define FHE_ASYNC_LDS 0
#endif

__device__ __forceinline__ void fhe_wait_async0() {
#if __has_builtin(__builtin_amdgcn_s_wait_asynccnt)
    __builtin_amdgcn_s_wait_asynccnt(0);
#else
    asm volatile("s_wait_asynccnt 0x0" ::: "memory");
#endif
}

// ---------------------------------------------------------------------------
// Embedding gather: h[i,:] = emb[x[i],:] * 0.01  (f32 -> f16)
// ---------------------------------------------------------------------------
__global__ __launch_bounds__(256) void fhe_embed_f16(const int* __restrict__ x,
                                                     const float* __restrict__ emb,
                                                     _Float16* __restrict__ h)
{
    const int row = blockIdx.x;                    // 0..8191  (B*T)
    const long tok = (long)x[row];
    const float* src = emb + tok * 1024;
    _Float16* dst = h + (long)row * 1024;
    const int d = threadIdx.x * 4;                 // 256 threads * 4 = 1024
    v4f f = *(const v4f*)(src + d);
    dst[d + 0] = (_Float16)(f.x * 0.01f);
    dst[d + 1] = (_Float16)(f.y * 0.01f);
    dst[d + 2] = (_Float16)(f.z * 0.01f);
    dst[d + 3] = (_Float16)(f.w * 0.01f);
}

// ---------------------------------------------------------------------------
// Tiled WMMA GEMM:  C = epilogue( alpha * A @ op(B) )
//   A : f16, row-major [M,K] (lda = K), optionally batched via strideA
//   B : if B_NK: [N,K] row-major (we compute A @ B^T), f32 or f16 (BT)
//       else   : [K,N] row-major f16 (we compute A @ B)
//   C : CT (f16 or f32), row-major [M,N]
//   ACT: fhe polynomial 0.1*v^2 + 0.1*v applied after alpha scaling
// Block: 256 threads = 8 wave32; tile BM x BN; each wave computes 32x64.
// Pipeline: double-buffered LDS; stage tile t+1 while computing tile t.
// ---------------------------------------------------------------------------
template<typename BT, bool B_NK, bool ACT, typename CT>
__global__ __launch_bounds__(256) void fhe_gemm_wmma(const _Float16* __restrict__ Ag,
                                                     const BT* __restrict__ Bg,
                                                     CT* __restrict__ Cg,
                                                     int M, int N, int K, float alpha,
                                                     long strideA, long strideB, long strideC)
{
    __shared__ _Float16 sA[2][BM * LDT];
    __shared__ _Float16 sB[2][BN * LDT];

    const int tid   = threadIdx.x;
    const int lane  = tid & 31;
    const int wave  = tid >> 5;        // 0..7
    const int waveM = wave & 3;        // 4 waves along M (32 rows each)
    const int waveN = wave >> 2;       // 2 waves along N (64 cols each)
    const int khalf = lane >> 4;       // 0 or 1 (lane half)
    const int l16   = lane & 15;

    const long bz = blockIdx.z;
    const _Float16* A = Ag + bz * strideA;
    const BT*       B = Bg + bz * strideB;
    CT*             C = Cg + bz * strideC;

    const int tileM = blockIdx.y * BM;
    const int tileN = blockIdx.x * BN;

    v8f acc[2][4];
#pragma unroll
    for (int mt = 0; mt < 2; ++mt)
#pragma unroll
        for (int nt = 0; nt < 4; ++nt)
#pragma unroll
            for (int i = 0; i < 8; ++i) acc[mt][nt][i] = 0.0f;

    // staging coordinates
    const int sRow = tid >> 1;               // 0..127 (row-pair scheme)
    const int sCol = (tid & 1) * 32;         // 0 or 32
    const int kr   = tid >> 2;               // 0..63   (!B_NK transpose scheme)
    const int n0   = (tid & 3) * 32;         // 0,32,64,96

    // register staging buffers (fallback / f32-convert / transpose paths)
    v8h aReg[4];
    v4f bRegF[8];
    v8h bRegH[4];

    // ---- issue tile-k0 global traffic into buffer `buf` --------------------
    auto issue = [&](int k0, int buf) {
        // A tile: BM x BKT f16
        const _Float16* ga = A + (long)(tileM + sRow) * K + k0 + sCol;
#if FHE_ASYNC_LDS
        {
            _Float16* la = &sA[buf][sRow * LDT + sCol];
            ASYNC_B128(ga + 0,  la + 0);
            ASYNC_B128(ga + 8,  la + 8);
            ASYNC_B128(ga + 16, la + 16);
            ASYNC_B128(ga + 24, la + 24);
        }
#else
        aReg[0] = *(const v8h*)(ga + 0);
        aReg[1] = *(const v8h*)(ga + 8);
        aReg[2] = *(const v8h*)(ga + 16);
        aReg[3] = *(const v8h*)(ga + 24);
#endif
        if constexpr (B_NK) {
            if constexpr (sizeof(BT) == 4) {
                // f32 weights [N,K]: load now, convert+store at commit
                const float* gb = (const float*)B + (long)(tileN + sRow) * K + k0 + sCol;
#pragma unroll
                for (int j = 0; j < 8; ++j)
                    bRegF[j] = *(const v4f*)(gb + 4 * j);
                __builtin_prefetch(gb + BKT, 0, 1);    // global_prefetch_b8 (next tile)
            } else {
                const _Float16* gb = (const _Float16*)B + (long)(tileN + sRow) * K + k0 + sCol;
#if FHE_ASYNC_LDS
                _Float16* lb = &sB[buf][sRow * LDT + sCol];
                ASYNC_B128(gb + 0,  lb + 0);
                ASYNC_B128(gb + 8,  lb + 8);
                ASYNC_B128(gb + 16, lb + 16);
                ASYNC_B128(gb + 24, lb + 24);
#else
                bRegH[0] = *(const v8h*)(gb + 0);
                bRegH[1] = *(const v8h*)(gb + 8);
                bRegH[2] = *(const v8h*)(gb + 16);
                bRegH[3] = *(const v8h*)(gb + 24);
#endif
            }
        } else {
            // B is [K,N] f16: coalesced read along N; transposed scatter at commit
            const _Float16* gb = (const _Float16*)B + (long)(k0 + kr) * N + tileN + n0;
#pragma unroll
            for (int j = 0; j < 4; ++j)
                bRegH[j] = *(const v8h*)(gb + 8 * j);
        }
    };

    // ---- commit staged data to LDS buffer `buf`, then barrier --------------
    auto commit = [&](int buf) {
#if !FHE_ASYNC_LDS
        {
            _Float16* la = &sA[buf][sRow * LDT + sCol];
            *(v8h*)(la + 0)  = aReg[0];
            *(v8h*)(la + 8)  = aReg[1];
            *(v8h*)(la + 16) = aReg[2];
            *(v8h*)(la + 24) = aReg[3];
        }
#endif
        if constexpr (B_NK) {
            if constexpr (sizeof(BT) == 4) {
                _Float16* lb = &sB[buf][sRow * LDT + sCol];
#pragma unroll
                for (int j = 0; j < 8; ++j) {
                    lb[4 * j + 0] = (_Float16)bRegF[j].x;
                    lb[4 * j + 1] = (_Float16)bRegF[j].y;
                    lb[4 * j + 2] = (_Float16)bRegF[j].z;
                    lb[4 * j + 3] = (_Float16)bRegF[j].w;
                }
            } else {
#if !FHE_ASYNC_LDS
                _Float16* lb = &sB[buf][sRow * LDT + sCol];
                *(v8h*)(lb + 0)  = bRegH[0];
                *(v8h*)(lb + 8)  = bRegH[1];
                *(v8h*)(lb + 16) = bRegH[2];
                *(v8h*)(lb + 24) = bRegH[3];
#endif
            }
        } else {
#pragma unroll
            for (int j = 0; j < 4; ++j)
#pragma unroll
                for (int e = 0; e < 8; ++e)
                    sB[buf][(n0 + 8 * j + e) * LDT + kr] = bRegH[j][e];
        }
#if FHE_ASYNC_LDS
        fhe_wait_async0();
#endif
        __syncthreads();
    };

    // ---- compute 16 WMMAs on LDS buffer `buf` ------------------------------
    auto compute = [&](int buf) {
        const _Float16* pA = &sA[buf][0];
        const _Float16* pB = &sB[buf][0];
#pragma unroll
        for (int kk = 0; kk < BKT; kk += 32) {
            // A frags: lane(row=l16, khalf) holds K = khalf*8+[0..7], 16+khalf*8+[0..7]
            v16h afrag[2];
#pragma unroll
            for (int mt = 0; mt < 2; ++mt) {
                const int row = waveM * 32 + mt * 16 + l16;
                const _Float16* ap = pA + row * LDT + kk + khalf * 8;
                v8h lo = *(const v8h*)(ap);
                v8h hi = *(const v8h*)(ap + 16);
                afrag[mt] = __builtin_shufflevector(lo, hi,
                    0,1,2,3,4,5,6,7,8,9,10,11,12,13,14,15);
            }
            // B frags: lane(col=l16, khalf) holds K = khalf*8+[0..7], 16+khalf*8+[0..7]
            v16h bfrag[4];
#pragma unroll
            for (int nt = 0; nt < 4; ++nt) {
                const int col = waveN * 64 + nt * 16 + l16;
                const _Float16* bp = pB + col * LDT + kk + khalf * 8;
                v8h lo = *(const v8h*)(bp);
                v8h hi = *(const v8h*)(bp + 16);
                bfrag[nt] = __builtin_shufflevector(lo, hi,
                    0,1,2,3,4,5,6,7,8,9,10,11,12,13,14,15);
            }
#pragma unroll
            for (int mt = 0; mt < 2; ++mt)
#pragma unroll
                for (int nt = 0; nt < 4; ++nt)
                    acc[mt][nt] = __builtin_amdgcn_wmma_f32_16x16x32_f16(
                        false, afrag[mt], false, bfrag[nt],
                        (short)0, acc[mt][nt], false, false);
        }
    };

    // ---- pipelined main loop ----------------------------------------------
    const int nk = K / BKT;
    issue(0, 0);
    commit(0);
    for (int t = 0; t < nk; ++t) {
        const int buf = t & 1;
        if (t + 1 < nk) issue((t + 1) * BKT, buf ^ 1);
        compute(buf);
        if (t + 1 < nk) commit(buf ^ 1);
    }

    // ---- epilogue: scale (+ poly activation) + store -----------------------
#pragma unroll
    for (int mt = 0; mt < 2; ++mt) {
#pragma unroll
        for (int nt = 0; nt < 4; ++nt) {
            const int col = tileN + waveN * 64 + nt * 16 + l16;
#pragma unroll
            for (int r = 0; r < 8; ++r) {
                const int row = tileM + waveM * 32 + mt * 16 + khalf * 8 + r;
                float v = acc[mt][nt][r] * alpha;
                if (ACT) v = v * v * 0.1f + v * 0.1f;   // fhe_act
                C[(long)row * N + col] = (CT)v;
            }
        }
    }
}

// ---------------------------------------------------------------------------
// Host-side pipeline
// ---------------------------------------------------------------------------
extern "C" void kernel_launch(void* const* d_in, const int* in_sizes, int n_in,
                              void* d_out, int out_size, void* d_ws, size_t ws_size,
                              hipStream_t stream)
{
    (void)in_sizes; (void)n_in; (void)out_size; (void)ws_size;

    const int*   x    = (const int*)  d_in[0];   // [4,2048]
    const float* emb  = (const float*)d_in[1];   // [32000,1024]
    const float* wq   = (const float*)d_in[2];   // [1024,1024]   (N,K)
    const float* wk   = (const float*)d_in[3];
    const float* wv   = (const float*)d_in[4];
    const float* wo   = (const float*)d_in[5];
    const float* wout = (const float*)d_in[6];   // [32000,1024]  (N,K)
    float* logits     = (float*)d_out;           // [4,2048,32000]

    const long BT_ = 4L * 2048;      // 8192 rows
    const long D   = 1024;
    const long T   = 2048;

    _Float16* ws   = (_Float16*)d_ws;
    _Float16* h16  = ws;                         // 8192*1024
    _Float16* q16  = h16 + BT_ * D;              // 8192*1024
    _Float16* k16  = q16 + BT_ * D;              // 8192*1024
    _Float16* v16  = k16 + BT_ * D;              // 8192*1024
    _Float16* at16 = v16 + BT_ * D;              // 4*2048*2048
    _Float16* ov16  = h16;                       // reuse: h dead after q/k/v
    _Float16* out16 = q16;                       // reuse: q dead after scores

    dim3 blk(256);

    // 1) h = emb[x] * 0.01
    fhe_embed_f16<<<dim3(BT_), blk, 0, stream>>>(x, emb, h16);

    // 2-4) q/k/v projections: (h @ w^T) * s
    dim3 gProj(D / BN, BT_ / BM, 1);
    fhe_gemm_wmma<float, true, false, _Float16><<<gProj, blk, 0, stream>>>(
        h16, wq, q16, (int)BT_, (int)D, (int)D, 0.01f, 0, 0, 0);
    fhe_gemm_wmma<float, true, false, _Float16><<<gProj, blk, 0, stream>>>(
        h16, wk, k16, (int)BT_, (int)D, (int)D, 0.01f, 0, 0, 0);
    fhe_gemm_wmma<float, true, false, _Float16><<<gProj, blk, 0, stream>>>(
        h16, wv, v16, (int)BT_, (int)D, (int)D, 0.1f, 0, 0, 0);

    // 5) attn = fhe_act( (q @ k^T) * 0.01 )  per batch
    dim3 gAttn(T / BN, T / BM, 4);
    fhe_gemm_wmma<_Float16, true, true, _Float16><<<gAttn, blk, 0, stream>>>(
        q16, k16, at16, (int)T, (int)T, (int)D, 0.01f, T * D, T * D, T * T);

    // 6) ov = (attn @ v) * 0.01  per batch   (B is [K,N] row-major)
    dim3 gAV(D / BN, T / BM, 4);
    fhe_gemm_wmma<_Float16, false, false, _Float16><<<gAV, blk, 0, stream>>>(
        at16, v16, ov16, (int)T, (int)D, (int)T, 0.01f, T * T, T * D, T * D);

    // 7) out = (ov @ wo^T) * 0.1
    fhe_gemm_wmma<float, true, false, _Float16><<<gProj, blk, 0, stream>>>(
        ov16, wo, out16, (int)BT_, (int)D, (int)D, 0.1f, 0, 0, 0);

    // 8) logits = out @ wout^T    (f32 output, the big one: 8192 x 32000 x 1024)
    dim3 gLog(32000 / BN, BT_ / BM, 1);
    fhe_gemm_wmma<float, true, false, float><<<gLog, blk, 0, stream>>>(
        out16, wout, logits, (int)BT_, 32000, (int)D, 1.0f, 0, 0, 0);
}
